// Classifier_61942018342957
// MI455X (gfx1250) — compile-verified, matrix operand order checked by reading, not run on hardware
//
#include <hip/hip_runtime.h>

// ---------------------------------------------------------------------------
// Pairwise cosine similarity: out[i][j] = (Z_i . Y_j) / max(|Z_i||Y_j|, 1e-8)
// Z, Y: 4096x4096 fp32 row-major.  out: 4096x4096 fp32.
//
// MI455X plan:
//  * Compute-bound GEMM (715 FLOP/B) -> bf16 WMMA 16x16x32 with 3-term split
//    accumulation (hi*hi + hi*lo + lo*hi) for ~fp32 accuracy at bf16 rate.
//  * fp32->bf16 hi/lo split done ONCE in a pre-pass into d_ws planes.
//  * GEMM mainloop: double-buffered LDS, GLOBAL_LOAD_ASYNC_TO_LDS_B128
//    (ASYNCcnt-tracked) staging overlapped with WMMA compute.
// ---------------------------------------------------------------------------

typedef __attribute__((ext_vector_type(16))) __bf16 v16bf;
typedef __attribute__((ext_vector_type(8)))  __bf16 v8bf;
typedef __attribute__((ext_vector_type(4)))  __bf16 v4bf;
typedef __attribute__((ext_vector_type(8)))  float  v8f;
typedef __attribute__((ext_vector_type(4)))  float  v4f;
typedef __attribute__((ext_vector_type(4)))  int    v4i;

#define MATN 4096
#define KB   32
#define EPS  1e-8f
#define PLANE ((size_t)MATN * MATN)      // elements per bf16 plane

#if __has_builtin(__builtin_amdgcn_global_load_async_to_lds_b128)
#define HAVE_ASYNC_LDS 1
#else
#define HAVE_ASYNC_LDS 0
#endif

static __device__ __forceinline__ void wait_async0() {
#if HAVE_ASYNC_LDS
#if __has_builtin(__builtin_amdgcn_s_wait_asynccnt)
    __builtin_amdgcn_s_wait_asynccnt(0);
#else
    asm volatile("s_wait_asynccnt 0" ::: "memory");
#endif
#endif
}

#if HAVE_ASYNC_LDS
static __device__ __forceinline__ void async_cp16(const __bf16* g, __bf16* l) {
    // builtin signature (per hipcc diagnostic): (v4i AS1*, v4i AS3*, imm, imm)
    v4i* gg = (v4i*)g;   // strips const (generic AS)
    v4i* ll = (v4i*)l;
    __builtin_amdgcn_global_load_async_to_lds_b128(
        (__attribute__((address_space(1))) v4i*)gg,
        (__attribute__((address_space(3))) v4i*)ll,
        0, 0);
}
#endif

// round-to-nearest-even f32 -> bf16
static __device__ __forceinline__ unsigned short f32_to_bf16_bits(float v) {
    unsigned u = __builtin_bit_cast(unsigned, v);
    unsigned r = u + 0x7FFFu + ((u >> 16) & 1u);
    return (unsigned short)(r >> 16);
}

static __device__ __forceinline__ void split_bf16(float v, __bf16& hi, __bf16& lo) {
    unsigned short hb = f32_to_bf16_bits(v);
    hi = __builtin_bit_cast(__bf16, hb);
    float hf = __builtin_bit_cast(float, (unsigned)hb << 16);
    lo = __builtin_bit_cast(__bf16, f32_to_bf16_bits(v - hf));
}

// WMMA 16-bit operand layout from an LDS 16x32 bf16 tile (row stride KB):
//   lane<16 : row = lane,    K in [0,8) and [16,24)
//   lane>=16: row = lane-16, K in [8,16) and [24,32)
static __device__ __forceinline__ v16bf load_operand(const __bf16* base, int row, int lane) {
    const __bf16* p = base + row * KB + ((lane & 16) ? 8 : 0);
    v8bf a = *(const v8bf*)(p);
    v8bf b = *(const v8bf*)(p + 16);
    return __builtin_shufflevector(a, b, 0, 1, 2, 3, 4, 5, 6, 7,
                                         8, 9, 10, 11, 12, 13, 14, 15);
}

// ---------------------------------------------------------------------------
// Kernel 1: fp32 -> bf16 hi/lo plane split for both matrices (one shot).
// ---------------------------------------------------------------------------
__global__ __launch_bounds__(256) void cosim_split(const float* __restrict__ Z,
                                                   const float* __restrict__ Y,
                                                   __bf16* __restrict__ zhi,
                                                   __bf16* __restrict__ zlo,
                                                   __bf16* __restrict__ yhi,
                                                   __bf16* __restrict__ ylo) {
    size_t c = (size_t)blockIdx.x * 256 + threadIdx.x;      // 0 .. 2*PLANE/4-1
    const size_t CPM = PLANE / 4;                           // chunks per matrix
    bool isY = c >= CPM;
    size_t idx = (isY ? c - CPM : c) * 4;
    const float* src = isY ? Y : Z;
    __bf16* dhi = isY ? yhi : zhi;
    __bf16* dlo = isY ? ylo : zlo;

    v4f v = *(const v4f*)(src + idx);
    v4bf h, l;
    #pragma unroll
    for (int j = 0; j < 4; ++j) {
        __bf16 hb, lb;
        split_bf16(v[j], hb, lb);
        h[j] = hb; l[j] = lb;
    }
    *(v4bf*)(dhi + idx) = h;
    *(v4bf*)(dlo + idx) = l;
}

// ---------------------------------------------------------------------------
// Kernel 2: row L2 norms -> norms[0..4095]=|Z_i|, [4096..8191]=|Y_j|
// ---------------------------------------------------------------------------
__global__ __launch_bounds__(256) void cosim_row_norms(const float* __restrict__ Z,
                                                       const float* __restrict__ Y,
                                                       float* __restrict__ norms) {
    int gw   = blockIdx.x * 8 + (threadIdx.x >> 5);
    int lane = threadIdx.x & 31;
    const float* src = (gw < MATN) ? Z : Y;
    int row = (gw < MATN) ? gw : (gw - MATN);

    const v4f* p = (const v4f*)(src + (size_t)row * MATN);
    float s = 0.f;
    #pragma unroll 4
    for (int i = lane; i < MATN / 4; i += 32) {
        v4f v = p[i];
        s = fmaf(v[0], v[0], s);
        s = fmaf(v[1], v[1], s);
        s = fmaf(v[2], v[2], s);
        s = fmaf(v[3], v[3], s);
    }
    #pragma unroll
    for (int off = 16; off >= 1; off >>= 1)
        s += __shfl_xor(s, off, 32);
    if (lane == 0) norms[gw] = sqrtf(s);
}

// ---------------------------------------------------------------------------
// Kernel 3: double-buffered async-staged bf16x3 WMMA GEMM + cosine epilogue.
// Block = 256 threads (8 waves, 4x2 grid), block tile 128x128.
// Wave tile = 32(M) x 64(N) = 2x4 accum tiles.
// LDS: 2 buffers x 4 planes(zhi,zlo,yhi,ylo) x 128x32 bf16 = 64 KB.
// ---------------------------------------------------------------------------
__global__ __launch_bounds__(256) void cosim_gemm_bf16x3(const __bf16* __restrict__ zhi,
                                                         const __bf16* __restrict__ zlo,
                                                         const __bf16* __restrict__ yhig,
                                                         const __bf16* __restrict__ ylog,
                                                         const float* __restrict__ norms,
                                                         float* __restrict__ out) {
    __shared__ __bf16 lds[2][4 * 128 * KB];

    const int tid  = threadIdx.x;
    const int lane = tid & 31;
    const int wave = tid >> 5;
    const int wr   = wave >> 1;
    const int wc   = wave & 1;
    const int lm   = lane & 15;
    const int blockM = blockIdx.y * 128;
    const int blockN = blockIdx.x * 128;

    const __bf16* planes[4] = { zhi, zlo, yhig, ylog };

    v8f acc[2][4];
    #pragma unroll
    for (int mt = 0; mt < 2; ++mt)
        #pragma unroll
        for (int nt = 0; nt < 4; ++nt)
            acc[mt][nt] = (v8f)(0.f);

    // Stage one 4-plane tile set (4 x 128x32 bf16 = 32 KB) into lds[buf].
    // 512 x 16B chunks per plane => 2 rounds of 256 threads.
    auto stage = [&](int buf, int k0) {
        #pragma unroll
        for (int p = 0; p < 4; ++p) {
            const int rowbase = (p < 2) ? blockM : blockN;
            #pragma unroll
            for (int q = 0; q < 2; ++q) {
                const int c   = tid + q * 256;          // 0..511
                const int r   = c >> 2;                 // 0..127
                const int col = (c & 3) << 3;           // 0,8,16,24
                const __bf16* g = planes[p] + (size_t)(rowbase + r) * MATN + k0 + col;
                __bf16* l = &lds[buf][p * (128 * KB) + r * KB + col];
#if HAVE_ASYNC_LDS
                async_cp16(g, l);
#else
                *(v8bf*)l = *(const v8bf*)g;            // VGPR bounce fallback
#endif
            }
        }
    };

    int cur = 0;
    stage(cur, 0);

    for (int k0 = 0; k0 < MATN; k0 += KB) {
        wait_async0();          // my fills of lds[cur] complete
        __syncthreads();        // everyone's fills visible; prior compute done

        if (k0 + KB < MATN)
            stage(cur ^ 1, k0 + KB);   // overlap next-chunk copy with compute

        const __bf16* xh = &lds[cur][0];
        const __bf16* xl = &lds[cur][128 * KB];
        const __bf16* yh = &lds[cur][2 * 128 * KB];
        const __bf16* yl = &lds[cur][3 * 128 * KB];

        v16bf Ah[2], Al[2], Bh[4], Bl[4];
        #pragma unroll
        for (int mt = 0; mt < 2; ++mt) {
            int r = wr * 32 + mt * 16 + lm;
            Ah[mt] = load_operand(xh, r, lane);
            Al[mt] = load_operand(xl, r, lane);
        }
        #pragma unroll
        for (int nt = 0; nt < 4; ++nt) {
            int cidx = wc * 64 + nt * 16 + lm;
            Bh[nt] = load_operand(yh, cidx, lane);
            Bl[nt] = load_operand(yl, cidx, lane);
        }

        #pragma unroll
        for (int mt = 0; mt < 2; ++mt) {
            #pragma unroll
            for (int nt = 0; nt < 4; ++nt) {
                acc[mt][nt] = __builtin_amdgcn_wmma_f32_16x16x32_bf16(
                    false, Ah[mt], false, Bh[nt], (short)0, acc[mt][nt], false, false);
                acc[mt][nt] = __builtin_amdgcn_wmma_f32_16x16x32_bf16(
                    false, Ah[mt], false, Bl[nt], (short)0, acc[mt][nt], false, false);
                acc[mt][nt] = __builtin_amdgcn_wmma_f32_16x16x32_bf16(
                    false, Al[mt], false, Bh[nt], (short)0, acc[mt][nt], false, false);
            }
        }
        cur ^= 1;
    }

    // Epilogue: C/D layout VGPR v -> M = v + 8*(lane>=16); lane%16 -> N.
    const float* nx = norms;
    const float* ny = norms + MATN;
    const int rhalf = (lane & 16) ? 8 : 0;
    #pragma unroll
    for (int mt = 0; mt < 2; ++mt) {
        float nxv[8];
        #pragma unroll
        for (int v = 0; v < 8; ++v)
            nxv[v] = nx[blockM + wr * 32 + mt * 16 + rhalf + v];
        #pragma unroll
        for (int nt = 0; nt < 4; ++nt) {
            int col = blockN + wc * 64 + nt * 16 + lm;
            float nyv = ny[col];
            #pragma unroll
            for (int v = 0; v < 8; ++v) {
                int row = blockM + wr * 32 + mt * 16 + rhalf + v;
                float denom = fmaxf(nxv[v] * nyv, EPS);
                out[(size_t)row * MATN + col] = acc[mt][nt][v] / denom;
            }
        }
    }
}

// ---------------------------------------------------------------------------
extern "C" void kernel_launch(void* const* d_in, const int* in_sizes, int n_in,
                              void* d_out, int out_size, void* d_ws, size_t ws_size,
                              hipStream_t stream) {
    const float* Z = (const float*)d_in[0];
    const float* Y = (const float*)d_in[1];
    float* out = (float*)d_out;

    // Workspace layout: [ norms: 8192 f32 (32KB) | zhi | zlo | yhi | ylo ]
    float*  norms = (float*)d_ws;
    __bf16* base  = (__bf16*)((char*)d_ws + 32768);
    __bf16* zhi = base;
    __bf16* zlo = base + PLANE;
    __bf16* yhi = base + 2 * PLANE;
    __bf16* ylo = base + 3 * PLANE;

    // 1) one-shot fp32 -> bf16 hi/lo split of both matrices
    cosim_split<<<(unsigned)(2 * PLANE / 4 / 256), 256, 0, stream>>>(Z, Y, zhi, zlo, yhi, ylo);

    // 2) row norms (reads original fp32 data)
    cosim_row_norms<<<8192 / 8, 256, 0, stream>>>(Z, Y, norms);

    // 3) WMMA GEMM + cosine normalization
    dim3 grid(MATN / 128, MATN / 128);
    cosim_gemm_bf16x3<<<grid, 256, 0, stream>>>(zhi, zlo, yhi, ylo, norms, out);
}
